// FlashAttention_18717467476234
// MI455X (gfx1250) — matrix operand order repaired and checked by reading, not compiled, not run on hardware
//
#include <hip/hip_runtime.h>
#include <math.h>
#include <stdint.h>

// ---------------------------------------------------------------------------
// Multi-head attention, fp32 in/out, f16 WMMA internally (fp32 accumulate).
//   D_MODEL=1024, HEADS=16, DK=64, B=4, S=2048  -> rows M = 8192
// Pipeline:
//   cvt (fp32->f16 once) -> qkv_proj (WMMA GEMM, async-LDS double buffer)
//   -> flash attention (WMMA, async-LDS double buffer, V pre-transposed)
//   -> out_proj (WMMA GEMM, fp32 epilogue)
// ---------------------------------------------------------------------------

#define D_MODEL 1024
#define SEQ     2048
#define BATCH   4
#define HEADS   16
#define DK      64
#define ROWS    (BATCH * SEQ)             // 8192
#define WMAT    (D_MODEL * D_MODEL)       // 1M elements per weight matrix

typedef __attribute__((ext_vector_type(16))) _Float16 v16h;
typedef __attribute__((ext_vector_type(8)))  _Float16 v8h;
typedef __attribute__((ext_vector_type(4)))  _Float16 v4h;
typedef __attribute__((ext_vector_type(8)))  float    v8f;
typedef __attribute__((ext_vector_type(4)))  int      int4v;

union F16Frag { v16h v; v8h h[2]; };

__device__ __forceinline__ v8f wmma_f16(v16h a, v16h b, v8f c) {
    // (neg_a, A, neg_b, B, c_mod, C, reuse_a, reuse_b)
    return __builtin_amdgcn_wmma_f32_16x16x32_f16(false, a, false, b,
                                                  (short)0, c, false, false);
}

__device__ __forceinline__ v16h ld_frag(const _Float16* p, int o0, int o1) {
    F16Frag f;
    f.h[0] = *(const v8h*)(p + o0);
    f.h[1] = *(const v8h*)(p + o1);
    return f.v;
}

// ---- CDNA5 async global->LDS copy (ASYNCcnt-tracked DMA, no VGPR staging) --
// Builtin signature (from toolchain diagnostic): param1 = int4* in AS(1)
// (global/__device__), param2 = LDS dest, then two int immediates.
__device__ __forceinline__ void async_copy_b128(void* lds, const void* gptr) {
#if __has_builtin(__builtin_amdgcn_global_load_async_to_lds_b128)
    __builtin_amdgcn_global_load_async_to_lds_b128(
        (__attribute__((address_space(1))) int4v*)gptr,
        (__attribute__((address_space(3))) int4v*)lds, 0, 0);
#else
    unsigned l = (unsigned)(size_t)lds;   // low 32 bits of flat LDS addr
    asm volatile("global_load_async_to_lds_b128 %0, %1, off"
                 :: "v"(l), "v"(gptr) : "memory");
#endif
}

#if __has_builtin(__builtin_amdgcn_s_wait_asynccnt)
#define WAIT_ASYNC(n) __builtin_amdgcn_s_wait_asynccnt(n)
#else
#define WAIT_ASYNC(n) asm volatile("s_wait_asynccnt %0" :: "n"(n) : "memory")
#endif

// Stage a 64x64 f16 tile (row stride `gstride` halfs) into LDS [64][72].
// 512 x 16B units, 2 per thread, all via async DMA.
__device__ __forceinline__ void stage_tile72(_Float16 (*dst)[72],
                                             const _Float16* src,
                                             int gstride, int tid) {
#pragma unroll
    for (int i = 0; i < 2; ++i) {
        const int unit = tid * 2 + i;          // 0..511
        const int r = unit >> 3;               // 0..63
        const int c = (unit & 7) * 8;          // 0..56 halfs
        async_copy_b128(&dst[r][c], src + (size_t)r * gstride + c);
    }
}

// ---------------------------------------------------------------------------
// Kernel 0a/0b: one-time fp32 -> f16 conversion (x and the 4 weight matrices).
// ---------------------------------------------------------------------------
__global__ __launch_bounds__(256) void cvt_x_kernel(
    const float* __restrict__ src, _Float16* __restrict__ dst) {
    const size_t i = ((size_t)blockIdx.x * 256 + threadIdx.x) * 4;
    const float4 f = *(const float4*)(src + i);
    v4h h; h[0] = (_Float16)f.x; h[1] = (_Float16)f.y;
           h[2] = (_Float16)f.z; h[3] = (_Float16)f.w;
    *(v4h*)(dst + i) = h;
}

__global__ __launch_bounds__(256) void cvt_w_kernel(
    const float* __restrict__ w0, const float* __restrict__ w1,
    const float* __restrict__ w2, const float* __restrict__ w3,
    _Float16* __restrict__ dst) {
    const int m = blockIdx.y;
    const float* s = (m == 0) ? w0 : (m == 1) ? w1 : (m == 2) ? w2 : w3;
    const size_t i = ((size_t)blockIdx.x * 256 + threadIdx.x) * 4;
    const float4 f = *(const float4*)(s + i);
    v4h h; h[0] = (_Float16)f.x; h[1] = (_Float16)f.y;
           h[2] = (_Float16)f.z; h[3] = (_Float16)f.w;
    *(v4h*)(dst + (size_t)m * WMAT + i) = h;
}

// ---------------------------------------------------------------------------
// Kernel 1: fused QKV projection (f16 in, f16 out).  Y = X @ W^T + b.
// Grid: (ROWS/64, 3*D_MODEL/64) = (128, 48), 256 threads (8 waves).
// Q,K written [B*H, S, DK]; V written TRANSPOSED [B*H, DK, S] for attention.
// ---------------------------------------------------------------------------
__global__ __launch_bounds__(256) void qkv_proj_kernel(
    const _Float16* __restrict__ Xh, const _Float16* __restrict__ Wh,
    const float* __restrict__ Bq, const float* __restrict__ Bk,
    const float* __restrict__ Bv,
    _Float16* __restrict__ Qb, _Float16* __restrict__ Kb,
    _Float16* __restrict__ VbT)
{
    __shared__ _Float16 Xs[2][64][72];
    __shared__ _Float16 Ws[2][64][72];

    const int tid  = threadIdx.x;
    const int wave = tid >> 5, lane = tid & 31;
    const int l16  = lane & 15, hi = lane >> 4;
    const int ab   = hi ? 16 : 0;     // A-fragment inner-K base
    const int kb   = hi ? 16 : 0;     // B-fragment inner-K base

    const int m0  = blockIdx.x * 64;
    const int n0g = blockIdx.y * 64;
    const int mat = n0g >> 10;                 // 0=Q 1=K 2=V
    const int n0  = n0g & 1023;

    const _Float16* W   = Wh + (size_t)mat * WMAT;
    const float*   bias = (mat == 0) ? Bq : (mat == 1) ? Bk : Bv;

    const int mi  = wave & 3;
    const int ni0 = (wave >> 2) * 2;

    v8f acc[2] = {};
    const _Float16* xsrc = Xh + (size_t)m0 * D_MODEL;
    const _Float16* wsrc = W  + (size_t)n0 * D_MODEL;

    stage_tile72(Xs[0], xsrc, D_MODEL, tid);
    stage_tile72(Ws[0], wsrc, D_MODEL, tid);

    const int NIT = D_MODEL / 64;   // 16
    for (int it = 0; it < NIT; ++it) {
        const int cur = it & 1;
        if (it + 1 < NIT) {
            const int k0 = (it + 1) * 64;
            stage_tile72(Xs[cur ^ 1], xsrc + k0, D_MODEL, tid);
            stage_tile72(Ws[cur ^ 1], wsrc + k0, D_MODEL, tid);
            WAIT_ASYNC(4);           // own 4 ops for buf[cur] done (in-order)
        } else {
            WAIT_ASYNC(0);
        }
        __syncthreads();             // tile visible from all waves

        const _Float16* xr = &Xs[cur][mi * 16 + l16][0];
        const _Float16* w0 = &Ws[cur][(ni0 + 0) * 16 + l16][0];
        const _Float16* w1 = &Ws[cur][(ni0 + 1) * 16 + l16][0];
#pragma unroll
        for (int c = 0; c < 2; ++c) {           // two 32-wide K chunks
            const v16h a  = ld_frag(xr, c * 32 + ab, c * 32 + ab + 8);
            const v16h b0 = ld_frag(w0, c * 32 + kb, c * 32 + kb + 8);
            const v16h b1 = ld_frag(w1, c * 32 + kb, c * 32 + kb + 8);
            acc[0] = wmma_f16(a, b0, acc[0]);
            acc[1] = wmma_f16(a, b1, acc[1]);
        }
        __syncthreads();             // all reads done before DMA overwrites
    }

    // Epilogue: bias add, f16 store.
#pragma unroll
    for (int t = 0; t < 2; ++t) {
        const int ncol = n0 + (ni0 + t) * 16 + l16;   // 0..1023
        const int h = ncol >> 6, d = ncol & 63;
        const float bv = bias[ncol];
#pragma unroll
        for (int v = 0; v < 8; ++v) {
            const int m = m0 + mi * 16 + hi * 8 + v;
            const int b = m >> 11, s = m & 2047;
            const float val = acc[t][v] + bv;
            if (mat == 2) {   // V: transposed [B*H, DK, S]
                VbT[((size_t)(b * HEADS + h) * DK + d) * SEQ + s] = (_Float16)val;
            } else {
                _Float16* dst = (mat == 0) ? Qb : Kb;
                dst[((size_t)(b * HEADS + h) * SEQ + s) * DK + d] = (_Float16)val;
            }
        }
    }
}

// ---------------------------------------------------------------------------
// Kernel 2: flash attention per (b,h).  Grid: (SEQ/128, B*H) = (16, 64).
// 8 waves x 16 query rows; 64-key steps, double-buffered async staging.
// ---------------------------------------------------------------------------
__global__ __launch_bounds__(256) void attn_kernel(
    const _Float16* __restrict__ Qb, const _Float16* __restrict__ Kb,
    const _Float16* __restrict__ VbT, _Float16* __restrict__ Ob)
{
    __shared__ _Float16 Ks[2][64][72];     // [key][d]
    __shared__ _Float16 Vt[2][64][72];     // [d][key]
    __shared__ _Float16 Ps[8][16][64];     // per-wave P staging (C->A relayout)

    const int tid  = threadIdx.x;
    const int wave = tid >> 5, lane = tid & 31;
    const int l16  = lane & 15, hi = lane >> 4;
    const int ab   = hi ? 16 : 0;
    const int kb   = hi ? 16 : 0;

    const int bh = blockIdx.y;                    // b*16 + h
    const int q0 = blockIdx.x * 128 + wave * 16;
    const size_t headK = (size_t)bh * SEQ * DK;   // Q/K row-major
    const size_t headV = (size_t)bh * DK * SEQ;   // V transposed

    // Q A-fragments held in registers for the whole key loop.
    const _Float16* qr = Qb + headK + (size_t)(q0 + l16) * DK;
    const v16h aQ0 = ld_frag(qr, ab, ab + 8);            // d 0..31
    const v16h aQ1 = ld_frag(qr, 32 + ab, 32 + ab + 8);  // d 32..63

    float mrow[8], lrow[8];
    v8f o[4] = {};
#pragma unroll
    for (int v = 0; v < 8; ++v) { mrow[v] = -INFINITY; lrow[v] = 0.0f; }

    const float scale = 0.125f;   // 1/sqrt(DK)
    const int NIT = SEQ / 64;     // 32

    stage_tile72(Ks[0], Kb + headK, DK, tid);
    stage_tile72(Vt[0], VbT + headV, SEQ, tid);

    for (int it = 0; it < NIT; ++it) {
        const int cur = it & 1;
        if (it + 1 < NIT) {
            const int kk = (it + 1) * 64;
            stage_tile72(Ks[cur ^ 1], Kb + headK + (size_t)kk * DK, DK, tid);
            stage_tile72(Vt[cur ^ 1], VbT + headV + kk, SEQ, tid);
            WAIT_ASYNC(4);
        } else {
            WAIT_ASYNC(0);
        }
        __syncthreads();

        // Scores: 4 key N-tiles of 16; inner d=64 in two 32-chunks.
        v8f sc[4];
#pragma unroll
        for (int t = 0; t < 4; ++t) {
            const _Float16* kr = &Ks[cur][t * 16 + l16][0]; // B[d][key]=K[key][d]
            v8f z = {};
            z = wmma_f16(aQ0, ld_frag(kr, kb, kb + 8), z);
            sc[t] = wmma_f16(aQ1, ld_frag(kr, 32 + kb, 32 + kb + 8), z);
        }

        // Online softmax (rows = accumulator elems, cols = lanes in 16-group).
        float alpha[8];
#pragma unroll
        for (int v = 0; v < 8; ++v) {
            const float s0 = sc[0][v] * scale, s1 = sc[1][v] * scale;
            const float s2 = sc[2][v] * scale, s3 = sc[3][v] * scale;
            float mx = fmaxf(fmaxf(s0, s1), fmaxf(s2, s3));
#pragma unroll
            for (int off = 1; off < 16; off <<= 1)
                mx = fmaxf(mx, __shfl_xor(mx, off, 32));
            const float mnew = fmaxf(mrow[v], mx);
            const float a  = __expf(mrow[v] - mnew);
            const float p0 = __expf(s0 - mnew), p1 = __expf(s1 - mnew);
            const float p2 = __expf(s2 - mnew), p3 = __expf(s3 - mnew);
            float ps = (p0 + p1) + (p2 + p3);
#pragma unroll
            for (int off = 1; off < 16; off <<= 1)
                ps += __shfl_xor(ps, off, 32);
            lrow[v] = lrow[v] * a + ps;
            mrow[v] = mnew;
            alpha[v] = a;
            _Float16* pw = &Ps[wave][hi * 8 + v][l16];
            pw[0]  = (_Float16)p0; pw[16] = (_Float16)p1;
            pw[32] = (_Float16)p2; pw[48] = (_Float16)p3;
        }
#pragma unroll
        for (int c = 0; c < 4; ++c)
#pragma unroll
            for (int v = 0; v < 8; ++v) o[c][v] *= alpha[v];
        __syncthreads();   // publish Ps (cross-lane) before A-fragment reads

        // O += P @ V : A = P (16x64), B = V (64 keys x 64 d) from Vt.
        const _Float16* pr = &Ps[wave][l16][0];
        const v16h aP0 = ld_frag(pr, ab, ab + 8);            // kk 0..31
        const v16h aP1 = ld_frag(pr, 32 + ab, 32 + ab + 8);  // kk 32..63
#pragma unroll
        for (int c = 0; c < 4; ++c) {
            const _Float16* vr = &Vt[cur][c * 16 + l16][0];  // B[kk][d]=Vt[d][kk]
            o[c] = wmma_f16(aP0, ld_frag(vr, kb, kb + 8), o[c]);
            o[c] = wmma_f16(aP1, ld_frag(vr, 32 + kb, 32 + kb + 8), o[c]);
        }
        __syncthreads();   // reads of Ks/Vt/Ps done before next DMA lands
    }

    // Epilogue: normalize, store f16 row-major [B, S, H*DK].
    const int b = bh >> 4, h = bh & 15;
#pragma unroll
    for (int v = 0; v < 8; ++v) {
        const float inv = 1.0f / lrow[v];
        const int q = q0 + hi * 8 + v;
        const size_t rowoff = ((size_t)b * SEQ + q) * D_MODEL + h * DK;
#pragma unroll
        for (int c = 0; c < 4; ++c)
            Ob[rowoff + c * 16 + l16] = (_Float16)(o[c][v] * inv);
    }
}

// ---------------------------------------------------------------------------
// Kernel 3: output projection.  Y = O @ Wo^T + bo, fp32 result.
// Grid: (ROWS/64, D_MODEL/64) = (128, 16).
// ---------------------------------------------------------------------------
__global__ __launch_bounds__(256) void out_proj_kernel(
    const _Float16* __restrict__ A, const _Float16* __restrict__ Wo,
    const float* __restrict__ Bo, float* __restrict__ Y)
{
    __shared__ _Float16 As[2][64][72];
    __shared__ _Float16 Ws[2][64][72];

    const int tid  = threadIdx.x;
    const int wave = tid >> 5, lane = tid & 31;
    const int l16  = lane & 15, hi = lane >> 4;
    const int ab   = hi ? 16 : 0;
    const int kb   = hi ? 16 : 0;

    const int m0 = blockIdx.x * 64;
    const int n0 = blockIdx.y * 64;
    const int mi  = wave & 3;
    const int ni0 = (wave >> 2) * 2;

    v8f acc[2] = {};
    const _Float16* asrc = A  + (size_t)m0 * D_MODEL;
    const _Float16* wsrc = Wo + (size_t)n0 * D_MODEL;

    stage_tile72(As[0], asrc, D_MODEL, tid);
    stage_tile72(Ws[0], wsrc, D_MODEL, tid);

    const int NIT = D_MODEL / 64;
    for (int it = 0; it < NIT; ++it) {
        const int cur = it & 1;
        if (it + 1 < NIT) {
            const int k0 = (it + 1) * 64;
            stage_tile72(As[cur ^ 1], asrc + k0, D_MODEL, tid);
            stage_tile72(Ws[cur ^ 1], wsrc + k0, D_MODEL, tid);
            WAIT_ASYNC(4);
        } else {
            WAIT_ASYNC(0);
        }
        __syncthreads();

        const _Float16* xr = &As[cur][mi * 16 + l16][0];
        const _Float16* w0 = &Ws[cur][(ni0 + 0) * 16 + l16][0];
        const _Float16* w1 = &Ws[cur][(ni0 + 1) * 16 + l16][0];
#pragma unroll
        for (int c = 0; c < 2; ++c) {
            const v16h a  = ld_frag(xr, c * 32 + ab, c * 32 + ab + 8);
            const v16h b0 = ld_frag(w0, c * 32 + kb, c * 32 + kb + 8);
            const v16h b1 = ld_frag(w1, c * 32 + kb, c * 32 + kb + 8);
            acc[0] = wmma_f16(a, b0, acc[0]);
            acc[1] = wmma_f16(a, b1, acc[1]);
        }
        __syncthreads();
    }

#pragma unroll
    for (int t = 0; t < 2; ++t) {
        const int n = n0 + (ni0 + t) * 16 + l16;
        const float bv = Bo[n];
#pragma unroll
        for (int v = 0; v < 8; ++v) {
            const int m = m0 + mi * 16 + hi * 8 + v;
            Y[(size_t)m * D_MODEL + n] = acc[t][v] + bv;
        }
    }
}

// ---------------------------------------------------------------------------
extern "C" void kernel_launch(void* const* d_in, const int* in_sizes, int n_in,
                              void* d_out, int out_size, void* d_ws,
                              size_t ws_size, hipStream_t stream)
{
    const float* x  = (const float*)d_in[0];
    const float* wq = (const float*)d_in[1];
    const float* bq = (const float*)d_in[2];
    const float* wk = (const float*)d_in[3];
    const float* bk = (const float*)d_in[4];
    const float* wv = (const float*)d_in[5];
    const float* bv = (const float*)d_in[6];
    const float* wo = (const float*)d_in[7];
    const float* bo = (const float*)d_in[8];

    const size_t XH_BYTES  = (size_t)ROWS * D_MODEL * sizeof(_Float16); // 16 MiB
    const size_t WH_BYTES  = (size_t)4 * WMAT * sizeof(_Float16);       //  8 MiB
    const size_t QKV_BYTES = (size_t)BATCH * HEADS * SEQ * DK * sizeof(_Float16);

    char* ws = (char*)d_ws;
    _Float16* Xh  = (_Float16*)(ws);
    _Float16* Wh  = (_Float16*)(ws + XH_BYTES);
    _Float16* Qb  = (_Float16*)(ws + XH_BYTES + WH_BYTES);
    _Float16* Kb  = (_Float16*)(ws + XH_BYTES + WH_BYTES + 1 * QKV_BYTES);
    _Float16* VbT = (_Float16*)(ws + XH_BYTES + WH_BYTES + 2 * QKV_BYTES);
    _Float16* Ob  = (_Float16*)(ws + XH_BYTES + WH_BYTES + 3 * QKV_BYTES);

    cvt_x_kernel<<<dim3(ROWS * D_MODEL / 1024), 256, 0, stream>>>(x, Xh);
    cvt_w_kernel<<<dim3(WMAT / 1024, 4), 256, 0, stream>>>(wq, wk, wv, wo, Wh);

    qkv_proj_kernel<<<dim3(ROWS / 64, 3 * D_MODEL / 64), 256, 0, stream>>>(
        Xh, Wh, bq, bk, bv, Qb, Kb, VbT);

    attn_kernel<<<dim3(SEQ / 128, BATCH * HEADS), 256, 0, stream>>>(
        Qb, Kb, VbT, Ob);

    out_proj_kernel<<<dim3(ROWS / 64, D_MODEL / 64), 256, 0, stream>>>(
        Ob, Wh + (size_t)3 * WMAT, bo, (float*)d_out);
}
